// ScaledDotProduct_3281355014500
// MI455X (gfx1250) — matrix operand order repaired and checked by reading, not compiled, not run on hardware
//
#include <hip/hip_runtime.h>

typedef __bf16 bf16;
typedef __attribute__((ext_vector_type(16))) __bf16 v16bf;
typedef __attribute__((ext_vector_type(8)))  __bf16 v8bf;
typedef __attribute__((ext_vector_type(8)))  float   v8f;
typedef __attribute__((ext_vector_type(4)))  unsigned int u32x4;
typedef __attribute__((ext_vector_type(8)))  int   i32x8;
typedef __attribute__((ext_vector_type(4)))  int   i32x4;

namespace {

constexpr int   kB = 2, kH = 8, kS = 4096, kD = 64;
constexpr float kInvT    = 0.125f;     // 1 / temperature(8.0)
constexpr float kMaskNeg = -3.0e38f;   // finite -inf stand-in (avoids NaN in online softmax)
constexpr int   kWaves      = 8;       // 256 threads / wave32
constexpr int   kQRowsPerWG = kWaves * 16;   // 128
constexpr int   kChunk      = 32;            // K/V rows staged per iteration
constexpr int   kNChunk     = kS / kChunk;   // 128

__device__ inline v8f wmma_bf16(v16bf a, v16bf b, v8f c) {
  // v_wmma_f32_16x16x32_bf16  (probe-confirmed 8-arg signature)
  return __builtin_amdgcn_wmma_f32_16x16x32_bf16(false, a, false, b, (short)0, c,
                                                 false, false);
}

__device__ inline v16bf cat8(v8bf a, v8bf b) {
  return __builtin_shufflevector(a, b, 0, 1, 2, 3, 4, 5, 6, 7,
                                       8, 9, 10, 11, 12, 13, 14, 15);
}

// 16 contiguous f32 in LDS -> v16bf operand fragment
__device__ inline v16bf cvt16(const float* p) {
  float4 x0 = *(const float4*)(p + 0);
  float4 x1 = *(const float4*)(p + 4);
  float4 x2 = *(const float4*)(p + 8);
  float4 x3 = *(const float4*)(p + 12);
  v16bf t;
  t[0]=(bf16)x0.x;  t[1]=(bf16)x0.y;  t[2]=(bf16)x0.z;  t[3]=(bf16)x0.w;
  t[4]=(bf16)x1.x;  t[5]=(bf16)x1.y;  t[6]=(bf16)x1.z;  t[7]=(bf16)x1.w;
  t[8]=(bf16)x2.x;  t[9]=(bf16)x2.y;  t[10]=(bf16)x2.z; t[11]=(bf16)x2.w;
  t[12]=(bf16)x3.x; t[13]=(bf16)x3.y; t[14]=(bf16)x3.z; t[15]=(bf16)x3.w;
  return t;
}

// TDM: DMA a 32x64 f32 tile (one K chunk, row-major, contiguous rows) into LDS.
// D# per CDNA5 ISA ch.8: group0 = {ctl, lds_addr, global_addr, type=2},
// group1 = dims/strides (data_size=4B, 2-D tile), groups 2/3 zero (<=2D tensor).
__device__ inline void tdm_load_tile_f32(const float* gsrc, unsigned int lds_byte_off) {
  const unsigned long long ga = (unsigned long long)(uintptr_t)gsrc;
  u32x4 g0;
  g0[0] = 1u;                                   // count=1, user descriptor
  g0[1] = lds_byte_off;                         // lds_addr (bytes)
  g0[2] = (unsigned int)ga;                     // global_addr[31:0]
  g0[3] = (unsigned int)((ga >> 32) & 0x01FFFFFFu) | (2u << 30);  // addr[56:32] | type=2

  i32x8 g1 = (i32x8)0;
  g1[0] = (int)(2u << 16);                      // data_size = 2 (4 bytes)
  g1[1] = (int)((unsigned)kD << 16);            // tensor_dim0[15:0] = 64
  g1[2] = (int)((unsigned)kS << 16);            // tensor_dim0 hi=0 | tensor_dim1 lo = 4096
  g1[3] = (int)((unsigned)kD << 16);            // tensor_dim1 hi=0 | tile_dim0 = 64
  g1[4] = (int)(unsigned)kChunk;                // tile_dim1 = 32, tile_dim2 = 0
  g1[5] = (int)(unsigned)kD;                    // tensor_dim0_stride lo = 64
  g1[6] = 0;                                    // stride hi / dim1_stride lo (unused, 2D)
  g1[7] = 0;

  i32x4 z4 = (i32x4)0;
#if defined(__clang_major__) && (__clang_major__ >= 23)
  i32x8 z8 = (i32x8)0;
  __builtin_amdgcn_tensor_load_to_lds(g0, g1, z4, z4, z8, 0);
#else
  __builtin_amdgcn_tensor_load_to_lds(g0, g1, z4, z4, 0);
#endif
}

} // namespace

// Fused scaled-dot-product attention, two-pass flash style, full attn emitted.
// Wave-level tiling: wave -> 16 Q rows; WG -> 128 Q rows; chunk -> 32 K rows.
// K chunks are moved by the Tensor Data Mover (double-buffered, TENSORcnt),
// V chunks are staged manually into the exact WMMA B-operand bf16 layout.
__global__ __launch_bounds__(256)
void sdpa_wmma_kernel(const float* __restrict__ qg, const float* __restrict__ kg,
                      const float* __restrict__ vg, const int* __restrict__ mask,
                      float* __restrict__ outg, float* __restrict__ attng) {
  // K chunk raw f32, double buffered for the TDM pipeline (2 x 8 KB)
  __shared__ __align__(16) float lds_kraw[2][kChunk * kD];
  // V chunk pre-swizzled into WMMA B-operand layout: [ntile(4)][lane(32)][e(16)]
  __shared__ __align__(32) bf16 lds_v[4 * 32 * 16];            // 4 KB
  // per-wave probability tile, row-major bf16 (16 x 32)
  __shared__ __align__(32) bf16 lds_p[kWaves * 16 * kChunk];   // 8 KB

  const int tid  = threadIdx.x;
  const int wave = tid >> 5;
  const int lane = tid & 31;
  const int n    = lane & 15;   // column / N index within 16-wide tile
  const int h    = lane >> 4;   // lane half
  const int bh   = blockIdx.y;
  const int qbase = blockIdx.x * kQRowsPerWG + wave * 16;

  const float* qp = qg + (size_t)bh * kS * kD;
  const float* kp = kg + (size_t)bh * kS * kD;
  const float* vp = vg + (size_t)bh * kS * kD;

  // ---------- A operand: this wave's 16x64 Q tile (contiguous k order) ----------
  // lane holds row (lane&15); elems 0..15 -> d = c*32 + h*16 + e.
  v16bf qa[2];
  {
    const float* qrow = qp + (size_t)(qbase + n) * kD;
#pragma unroll
    for (int c = 0; c < 2; ++c) {
      const int d0 = c * 32 + h * 16;
      float4 a0 = *(const float4*)(qrow + d0);
      float4 a1 = *(const float4*)(qrow + d0 + 4);
      float4 b0 = *(const float4*)(qrow + d0 + 8);
      float4 b1 = *(const float4*)(qrow + d0 + 12);
      v16bf t;
      t[0]=(bf16)a0.x;  t[1]=(bf16)a0.y;  t[2]=(bf16)a0.z;  t[3]=(bf16)a0.w;
      t[4]=(bf16)a1.x;  t[5]=(bf16)a1.y;  t[6]=(bf16)a1.z;  t[7]=(bf16)a1.w;
      t[8]=(bf16)b0.x;  t[9]=(bf16)b0.y;  t[10]=(bf16)b0.z; t[11]=(bf16)b0.w;
      t[12]=(bf16)b1.x; t[13]=(bf16)b1.y; t[14]=(bf16)b1.z; t[15]=(bf16)b1.w;
      qa[c] = t;
    }
  }

  const unsigned int kraw_off0 = (unsigned int)(uintptr_t)&lds_kraw[0][0];
  const unsigned int kraw_off1 = (unsigned int)(uintptr_t)&lds_kraw[1][0];

  // ---------- Pass 1: online row max / sum over all 4096 columns ----------
  float rm[8], rs[8];
#pragma unroll
  for (int r = 0; r < 8; ++r) { rm[r] = kMaskNeg; rs[r] = 0.0f; }

  if (wave == 0) tdm_load_tile_f32(kp, kraw_off0);   // prologue: chunk 0

  for (int ck = 0; ck < kNChunk; ++ck) {
    const int kbase = ck * kChunk;
    __syncthreads();   // all waves done reading the buffer TDM will overwrite
    if (wave == 0) {
      if (ck + 1 < kNChunk) {
        tdm_load_tile_f32(kp + (size_t)(kbase + kChunk) * kD,
                          ((ck + 1) & 1) ? kraw_off1 : kraw_off0);
        __builtin_amdgcn_s_wait_tensorcnt(1);   // chunk ck complete
      } else {
        __builtin_amdgcn_s_wait_tensorcnt(0);
      }
    }
    __syncthreads();

    const float* kbuf = lds_kraw[ck & 1];
#pragma unroll
    for (int j = 0; j < 2; ++j) {
      const int row = j * 16 + n;   // K-row within chunk this lane serves as N
      v16bf kb0 = cvt16(&kbuf[row * kD + h * 16]);        // d 0..31
      v16bf kb1 = cvt16(&kbuf[row * kD + 32 + h * 16]);   // d 32..63
      v8f sc = {};
      sc = wmma_bf16(qa[0], kb0, sc);
      sc = wmma_bf16(qa[1], kb1, sc);
#pragma unroll
      for (int r = 0; r < 8; ++r) {
        const int m   = h * 8 + r;
        const int col = kbase + j * 16 + n;
        float x = sc[r] * kInvT;
        if (mask[(size_t)(qbase + m) * kS + col] == 0) x = kMaskNeg;
        const float nm = fmaxf(rm[r], x);
        rs[r] = rs[r] * __expf(rm[r] - nm) + __expf(x - nm);
        rm[r] = nm;
      }
    }
  }

  // cross-lane reduce within each 16-lane half (rows split by h in C layout)
  float rsi[8];
#pragma unroll
  for (int r = 0; r < 8; ++r) {
#pragma unroll
    for (int off = 8; off > 0; off >>= 1) {
      const float om = __shfl_xor(rm[r], off, 16);
      const float os = __shfl_xor(rs[r], off, 16);
      const float nm = fmaxf(rm[r], om);
      rs[r] = rs[r] * __expf(rm[r] - nm) + os * __expf(om - nm);
      rm[r] = nm;
    }
    rsi[r] = 1.0f / rs[r];
  }

  // ---------- Pass 2: write attn, accumulate out = attn @ V ----------
  v8f oc[4] = {v8f{}, v8f{}, v8f{}, v8f{}};

  __syncthreads();
  if (wave == 0) tdm_load_tile_f32(kp, kraw_off0);   // prologue: chunk 0 again

  for (int ck = 0; ck < kNChunk; ++ck) {
    const int kbase = ck * kChunk;
    __syncthreads();   // protects K buffer overwrite and lds_v reuse
    if (wave == 0) {
      if (ck + 1 < kNChunk) {
        tdm_load_tile_f32(kp + (size_t)(kbase + kChunk) * kD,
                          ((ck + 1) & 1) ? kraw_off1 : kraw_off0);
        __builtin_amdgcn_s_wait_tensorcnt(1);
      } else {
        __builtin_amdgcn_s_wait_tensorcnt(0);
      }
    }
    {   // V staging, swizzled into B-operand layout: elem e <-> K row (h2*16+e)
      const int d  = tid & 63;
      const int k0 = (tid >> 6) * 8;          // {0,8,16,24}
      const int h2 = k0 >> 4;                 // lane half by K row
      const int e0 = k0 & 15;                 // element base within lane
      const float* src = vp + (size_t)(kbase + k0) * kD + d;
      v8bf pk;
#pragma unroll
      for (int i = 0; i < 8; ++i) pk[i] = (bf16)src[(size_t)i * kD];
      const int t2 = d >> 4, nn = d & 15;
      *(v8bf*)&lds_v[((t2 * 32) + (h2 * 16 + nn)) * 16 + e0] = pk;
      __builtin_prefetch(src + kChunk * kD, 0, 0);   // global_prefetch_b8
    }
    __syncthreads();

    const float* kbuf = lds_kraw[ck & 1];
    bf16* pw = &lds_p[wave * 16 * kChunk];
#pragma unroll
    for (int j = 0; j < 2; ++j) {
      const int row = j * 16 + n;
      v16bf kb0 = cvt16(&kbuf[row * kD + h * 16]);
      v16bf kb1 = cvt16(&kbuf[row * kD + 32 + h * 16]);
      v8f sc = {};
      sc = wmma_bf16(qa[0], kb0, sc);
      sc = wmma_bf16(qa[1], kb1, sc);
#pragma unroll
      for (int r = 0; r < 8; ++r) {
        const int m   = h * 8 + r;
        const int col = kbase + j * 16 + n;
        float x = sc[r] * kInvT;
        if (mask[(size_t)(qbase + m) * kS + col] == 0) x = kMaskNeg;
        const float p = __expf(x - rm[r]) * rsi[r];
        attng[(size_t)bh * kS * kS + (size_t)(qbase + m) * kS + col] = p;
        pw[m * kChunk + j * 16 + n] = (bf16)p;   // stage for layout transpose
      }
    }

    // per-wave LDS ordering: all lanes' p stores complete before A' reads
    asm volatile("s_wait_dscnt 0x0" ::: "memory");

    // A' operand: row (lane&15), contiguous layout over the 32 chunk cols (k = h*16 + e)
    const bf16* prow = &pw[n * kChunk];
    v8bf a0 = *(const v8bf*)&prow[h * 16];
    v8bf a1 = *(const v8bf*)&prow[h * 16 + 8];
    v16bf pa = cat8(a0, a1);
#pragma unroll
    for (int t = 0; t < 4; ++t) {
      v16bf vb = *(const v16bf*)&lds_v[(t * 32 + lane) * 16];
      oc[t] = wmma_bf16(pa, vb, oc[t]);
    }
  }

  // ---------- store output tile (C layout -> global) ----------
#pragma unroll
  for (int t = 0; t < 4; ++t) {
#pragma unroll
    for (int r = 0; r < 8; ++r) {
      const int m = h * 8 + r;
      outg[(size_t)bh * kS * kD + (size_t)(qbase + m) * kD + t * 16 + n] = oc[t][r];
    }
  }
}

extern "C" void kernel_launch(void* const* d_in, const int* in_sizes, int n_in,
                              void* d_out, int out_size, void* d_ws, size_t ws_size,
                              hipStream_t stream) {
  (void)in_sizes; (void)n_in; (void)d_ws; (void)ws_size; (void)out_size;
  const float* q    = (const float*)d_in[0];
  const float* k    = (const float*)d_in[1];
  const float* v    = (const float*)d_in[2];
  const int*   mask = (const int*)d_in[3];

  float* out  = (float*)d_out;                               // [B,H,S,D]
  float* attn = out + (size_t)kB * kH * kS * kD;             // [B,H,S,S]

  dim3 grid(kS / kQRowsPerWG, kB * kH);  // (32, 16)
  sdpa_wmma_kernel<<<grid, kWaves * 32, 0, stream>>>(q, k, v, mask, out, attn);
}
